// MultiHeadSelfAttention_44796508897377
// MI455X (gfx1250) — compile-verified, matrix operand order checked by reading, not run on hardware
//
#include <hip/hip_runtime.h>

typedef __bf16 bf16;
typedef __attribute__((ext_vector_type(16))) __bf16 bf16x16;
typedef __attribute__((ext_vector_type(4)))  __bf16 bf16x4;
typedef __attribute__((ext_vector_type(8)))  float  f32x8;
typedef __attribute__((ext_vector_type(4)))  float  f32x4;
typedef __attribute__((ext_vector_type(4)))  unsigned int u32x4;
typedef __attribute__((ext_vector_type(8)))  int   i32x8;
typedef __attribute__((ext_vector_type(4)))  int   i32x4;

#define S_LEN 2048
#define D_DIM 1024
#define NH    16
#define HS    64

// TDM builtin arity: therock toolchain ships the TDM header (6-arg builtin);
// ROCm 7.2 (clang-22) is 5-arg.
#if defined(__has_include)
#if __has_include(<hip/amd_detail/amd_gfx1250_TDM.h>)
#define TDM_ARGS6 1
#else
#define TDM_ARGS6 0
#endif
#else
#define TDM_ARGS6 0
#endif

#if defined(__has_builtin)
#if __has_builtin(__builtin_amdgcn_tensor_load_to_lds)
#define HAVE_TDM 1
#else
#define HAVE_TDM 0
#endif
#else
#define HAVE_TDM 0
#endif

__device__ __forceinline__ f32x8 wmma_bf16(bf16x16 a, bf16x16 b, f32x8 c) {
    // 8 args: (neg_a, A, neg_b, B, c_mod, C, reuse_a, reuse_b)
    return __builtin_amdgcn_wmma_f32_16x16x32_bf16(false, a, false, b, (short)0, c, false, false);
}

#if HAVE_TDM
// ---------------------------------------------------------------------------
// TDM 2D tile load: tileRows x 64 bf16, row stride D_DIM, global->LDS.
// D# group0/group1 per ISA 8.3/8.4 (count=1, type=2, data_size=2B). The LDS
// aperture maps flat addr[31:0] to the LDS byte offset, so pointer truncation
// gives D#.lds_addr. Tracked on TENSORcnt.
// ---------------------------------------------------------------------------
__device__ __forceinline__ void tdm_load_2d(const bf16* gtile, const bf16* ltile,
                                            int tileRows) {
    const unsigned long long ga = (unsigned long long)gtile;
    const unsigned lo = (unsigned)(unsigned long long)ltile;  // LDS byte offset
    u32x4 g0;
    g0[0] = 1u;                                               // count=1 (valid user D#)
    g0[1] = lo;                                               // lds_addr
    g0[2] = (unsigned)ga;                                     // global_addr[31:0]
    g0[3] = (unsigned)((ga >> 32) & 0x01FFFFFFull) | (2u << 30); // addr[56:32] | type=2
    i32x8 g1;
    g1[0] = 1 << 16;                                          // data_size = 2 bytes
    g1[1] = (int)((D_DIM & 0xFFFF) << 16);                    // tensor_dim0[15:0]
    g1[2] = (int)((D_DIM >> 16) & 0xFFFF) | ((tileRows & 0xFFFF) << 16); // dim0 hi | tensor_dim1 lo
    g1[3] = (int)(64 << 16);                                  // tensor_dim1 hi | tile_dim0 = 64
    g1[4] = tileRows & 0xFFFF;                                // tile_dim1 (tile_dim2 = 0)
    g1[5] = D_DIM;                                            // tensor_dim0_stride[31:0]
    g1[6] = 0;
    g1[7] = 0;
    i32x4 z4 = {};
#if TDM_ARGS6
    i32x8 z8 = {};
    __builtin_amdgcn_tensor_load_to_lds(g0, g1, z4, z4, z8, 0);
#else
    __builtin_amdgcn_tensor_load_to_lds(g0, g1, z4, z4, 0);
#endif
}
#endif // HAVE_TDM

// ---------------------------------------------------------------------------
// fp32 -> bf16 elementwise conversion (vectorized by 4)
// ---------------------------------------------------------------------------
__global__ void cvt_f32_bf16(const float* __restrict__ src, bf16* __restrict__ dst, int n4) {
    int i = blockIdx.x * blockDim.x + threadIdx.x;
    if (i < n4) {
        f32x4 v = ((const f32x4*)src)[i];
        bf16x4 o = { (bf16)v.x, (bf16)v.y, (bf16)v.z, (bf16)v.w };
        ((bf16x4*)dst)[i] = o;
    }
}

// ---------------------------------------------------------------------------
// One K-step (K=64) of the per-wave 16x64 output tile: batch all fragment
// loads (2 A frags + 8 B frags, each contiguous 32B in LDS), one dscnt wait,
// then 8 back-to-back WMMAs round-robined over the 4 accumulators so
// same-accumulator ops are 4 apart.
// ---------------------------------------------------------------------------
__device__ __forceinline__ void gemm_step(const bf16* __restrict__ Asb,
                                          const bf16* __restrict__ Bsb,
                                          int wave, int l16, int half,
                                          f32x8 acc[4]) {
    const bf16* at = Asb + (wave * 16 + l16) * 64 + half * 16;
    const bf16* bt = Bsb + l16 * 64 + half * 16;
    const bf16x16 a0  = *(const bf16x16*)(at);
    const bf16x16 a1  = *(const bf16x16*)(at + 32);
    const bf16x16 b00 = *(const bf16x16*)(bt);
    const bf16x16 b01 = *(const bf16x16*)(bt + 32);
    const bf16x16 b10 = *(const bf16x16*)(bt + 16 * 64);
    const bf16x16 b11 = *(const bf16x16*)(bt + 16 * 64 + 32);
    const bf16x16 b20 = *(const bf16x16*)(bt + 32 * 64);
    const bf16x16 b21 = *(const bf16x16*)(bt + 32 * 64 + 32);
    const bf16x16 b30 = *(const bf16x16*)(bt + 48 * 64);
    const bf16x16 b31 = *(const bf16x16*)(bt + 48 * 64 + 32);
    acc[0] = wmma_bf16(a0, b00, acc[0]);
    acc[1] = wmma_bf16(a0, b10, acc[1]);
    acc[2] = wmma_bf16(a0, b20, acc[2]);
    acc[3] = wmma_bf16(a0, b30, acc[3]);
    acc[0] = wmma_bf16(a1, b01, acc[0]);
    acc[1] = wmma_bf16(a1, b11, acc[1]);
    acc[2] = wmma_bf16(a1, b21, acc[2]);
    acc[3] = wmma_bf16(a1, b31, acc[3]);
}

// ---------------------------------------------------------------------------
// GEMM: out[m,n] = sum_k A[m,k] * W[n,k]   (A: [M,K] rm, W: [N,K] rm, B = W^T)
// M = 4096, K = N = 1024. Block = 8 waves, tile M=128 x N=64, K-step 64.
// TDM double-buffered (waves 0-3 DMA A slices, waves 4-7 DMA W slices);
// K loop unrolled x2 with statically-addressed buffers to avoid accumulator
// phi copies. Protocol per phase: wait tensorcnt -> barrier -> issue next
// tile into the buffer last read one phase ago -> compute current buffer.
// mode 0: bf16 out, split-head [B,H,S,64]   (Q, K)
// mode 1: bf16 out, transposed [B,H,64,S]   (V^T)
// mode 2: f32  out, [B,S,D]                 (final O projection)
// ---------------------------------------------------------------------------
__global__ void __launch_bounds__(256) gemm_bf16_wmma(
        const bf16* __restrict__ A, const bf16* __restrict__ W,
        bf16* __restrict__ outb, float* __restrict__ outf, int mode) {
    __shared__ bf16 As[2][128 * 64];   // 32 KB
    __shared__ bf16 Bs[2][64 * 64];    // 16 KB
    const int lane = threadIdx.x & 31;
    const int wave = threadIdx.x >> 5;
    const int half = lane >> 4;        // K-segment selector (A/B frag layout)
    const int l16  = lane & 15;        // row (A frag) / column (B frag, C/D)
    const int m0 = blockIdx.y * 128;
    const int n0 = blockIdx.x * 64;

    f32x8 acc[4] = {};
    const int NT = D_DIM / 64;         // 16 K-steps

#if HAVE_TDM
    // Per-wave DMA slice assignment.
    const bool isA = (wave < 4);
    const int sliceRows = isA ? 32 : 16;
    const int srow = isA ? (wave * 32) : ((wave - 4) * 16);
    const bf16* gbase = isA ? (A + (size_t)(m0 + srow) * D_DIM)
                            : (W + (size_t)(n0 + srow) * D_DIM);
    const bf16* lbase0 = isA ? &As[0][srow * 64] : &Bs[0][srow * 64];
    const bf16* lbase1 = isA ? &As[1][srow * 64] : &Bs[1][srow * 64];

    tdm_load_2d(gbase, lbase0, sliceRows);                  // prologue: tile 0 -> buf0

    for (int kt = 0; kt < NT; kt += 2) {
        // ---- phase A: tile kt in buf0
        __builtin_amdgcn_s_wait_tensorcnt((short)0);        // my slice landed
        __syncthreads();                                    // all slices landed; buf1 reads done
        tdm_load_2d(gbase + (kt + 1) * 64, lbase1, sliceRows);
        gemm_step(&As[0][0], &Bs[0][0], wave, l16, half, acc);

        // ---- phase B: tile kt+1 in buf1
        __builtin_amdgcn_s_wait_tensorcnt((short)0);
        __syncthreads();                                    // buf0 reads done
        if (kt + 2 < NT)
            tdm_load_2d(gbase + (kt + 2) * 64, lbase0, sliceRows);
        gemm_step(&As[1][0], &Bs[1][0], wave, l16, half, acc);
    }
#else
    // Fallback: direct-from-global fragments (B columns == W rows, contiguous).
    const bf16* ap = A + (size_t)(m0 + wave * 16 + l16) * D_DIM + half * 16;
    const bf16* wp = W + (size_t)(n0 + l16) * D_DIM + half * 16;
    for (int k = 0; k < D_DIM; k += 32) {
        bf16x16 a = *(const bf16x16*)(ap + k);
#pragma unroll
        for (int t = 0; t < 4; ++t) {
            bf16x16 b = *(const bf16x16*)(wp + (size_t)(t * 16) * D_DIM + k);
            acc[t] = wmma_bf16(a, b, acc[t]);
        }
    }
#endif

    // C/D layout: VGPR r -> row r (lanes 0-15) and row r+8 (lanes 16-31), col = lane&15
#pragma unroll
    for (int t = 0; t < 4; ++t) {
#pragma unroll
        for (int r = 0; r < 8; ++r) {
            const int gm = m0 + wave * 16 + r + half * 8;
            const int gn = n0 + t * 16 + l16;
            const float v = acc[t][r];
            const int bb = gm >> 11;            // / S_LEN
            const int ss = gm & (S_LEN - 1);
            if (mode == 0) {
                const int hh = gn >> 6, cc = gn & 63;
                outb[(((size_t)(bb * NH + hh) * S_LEN + ss) << 6) + cc] = (bf16)v;
            } else if (mode == 1) {
                const int hh = gn >> 6, cc = gn & 63;
                outb[((size_t)(bb * NH + hh) * HS + cc) * S_LEN + ss] = (bf16)v;
            } else {
                outf[(size_t)gm * D_DIM + gn] = v;
            }
        }
    }
}

// ---------------------------------------------------------------------------
// Causal flash attention forward. One wave owns a 16-query tile, streams
// 32-key tiles with online softmax. Q/K in [B,H,S,64], V pre-transposed
// [B,H,64,S] so all WMMA B-fragments are contiguous 32B loads per lane.
// P (16x32 probs) goes through a per-wave LDS buffer to re-layout into an
// A-fragment for the PV WMMAs. Block = 4 waves, no cross-wave sync needed.
// ---------------------------------------------------------------------------
__global__ void attn_fwd_wmma(const bf16* __restrict__ Q, const bf16* __restrict__ Kc,
                              const bf16* __restrict__ Vt, bf16* __restrict__ ctx) {
    __shared__ bf16 Pbuf[4][16 * 32];
    const int lane = threadIdx.x & 31;
    const int wave = threadIdx.x >> 5;
    const int half = lane >> 4;
    const int l16  = lane & 15;
    const int h = blockIdx.y, b = blockIdx.z;
    const int q0 = (blockIdx.x * 4 + wave) * 16;

    const bf16* Qh = Q  + (size_t)(b * NH + h) * S_LEN * HS;
    const bf16* Kh = Kc + (size_t)(b * NH + h) * S_LEN * HS;
    const bf16* Vh = Vt + (size_t)(b * NH + h) * HS * S_LEN;

    // Q A-fragments for head-dim halves 0..31 and 32..63
    const bf16x16 a0 = *(const bf16x16*)(Qh + (size_t)(q0 + l16) * HS + half * 16);
    const bf16x16 a1 = *(const bf16x16*)(Qh + (size_t)(q0 + l16) * HS + 32 + half * 16);

    f32x8 acc[4] = {};
    float mrow[8], lrow[8];
#pragma unroll
    for (int r = 0; r < 8; ++r) { mrow[r] = -3.0e38f; lrow[r] = 0.0f; }
    const float scale = 0.125f;               // 1/sqrt(64)
    const int qbase = q0 + half * 8;
    bf16* P = &Pbuf[wave][0];

    for (int kb = 0; kb < q0 + 16; kb += 32) {
        // ---- scores: S = Q K^T (two 16-key subtiles, K-dim 64 = 2 chained WMMAs)
        f32x8 s[2];
#pragma unroll
        for (int t = 0; t < 2; ++t) {
            const bf16* kr = Kh + (size_t)(kb + t * 16 + l16) * HS + half * 16;
            bf16x16 b0 = *(const bf16x16*)kr;
            bf16x16 b1 = *(const bf16x16*)(kr + 32);
            f32x8 z = {};
            s[t] = wmma_bf16(a1, b1, wmma_bf16(a0, b0, z));
        }

        // ---- causal mask + online softmax (row stats redundant per 16-lane half)
        float alpha[8];
#pragma unroll
        for (int r = 0; r < 8; ++r) {
            const int qi = qbase + r;
#pragma unroll
            for (int t = 0; t < 2; ++t) {
                const int ki = kb + t * 16 + l16;
                const float v = s[t][r] * scale;
                s[t][r] = (ki > qi) ? -3.0e38f : v;
            }
            float mv = fmaxf(s[0][r], s[1][r]);
#pragma unroll
            for (int off = 1; off < 16; off <<= 1)
                mv = fmaxf(mv, __shfl_xor(mv, off, 32));
            const float mn = fmaxf(mrow[r], mv);
            alpha[r] = __expf(mrow[r] - mn);
            mrow[r] = mn;
            const float p0 = __expf(s[0][r] - mn);
            const float p1 = __expf(s[1][r] - mn);
            P[(r + half * 8) * 32 + l16]      = (bf16)p0;   // stage P into LDS
            P[(r + half * 8) * 32 + 16 + l16] = (bf16)p1;
            float sv = p0 + p1;
#pragma unroll
            for (int off = 1; off < 16; off <<= 1)
                sv += __shfl_xor(sv, off, 32);
            lrow[r] = lrow[r] * alpha[r] + sv;
        }

        // ---- rescale accumulator by exp(m_old - m_new)
#pragma unroll
        for (int n = 0; n < 4; ++n)
#pragma unroll
            for (int r = 0; r < 8; ++r)
                acc[n][r] *= alpha[r];

        // ---- PV: O += P V. Re-load P as an A-fragment (LDS cross-lane re-layout).
        asm volatile("s_wait_dscnt 0x0" ::: "memory");
        const bf16x16 pa = *(const bf16x16*)(P + l16 * 32 + half * 16);
#pragma unroll
        for (int n = 0; n < 4; ++n) {
            bf16x16 bv = *(const bf16x16*)(Vh + (size_t)(n * 16 + l16) * S_LEN + kb + half * 16);
            acc[n] = wmma_bf16(pa, bv, acc[n]);
        }
    }

    // ---- epilogue: normalize and write context in [B,S,D] (d = h*64 + c)
#pragma unroll
    for (int n = 0; n < 4; ++n) {
#pragma unroll
        for (int r = 0; r < 8; ++r) {
            const int qi = q0 + r + half * 8;
            const float v = acc[n][r] / lrow[r];
            ctx[(size_t)(b * S_LEN + qi) * D_DIM + h * HS + n * 16 + l16] = (bf16)v;
        }
    }
}

// ---------------------------------------------------------------------------
extern "C" void kernel_launch(void* const* d_in, const int* in_sizes, int n_in,
                              void* d_out, int out_size, void* d_ws, size_t ws_size,
                              hipStream_t stream) {
    (void)in_sizes; (void)n_in; (void)out_size; (void)ws_size;
    const float* x  = (const float*)d_in[0];
    const float* wq = (const float*)d_in[1];
    const float* wk = (const float*)d_in[2];
    const float* wv = (const float*)d_in[3];
    const float* wo = (const float*)d_in[4];

    const size_t nX = (size_t)2 * S_LEN * D_DIM;   // 4,194,304 elems
    const size_t nW = (size_t)D_DIM * D_DIM;       // 1,048,576 elems

    bf16* xb  = (bf16*)d_ws;        // bf16 x            [B,S,D]
    bf16* wqb = xb  + nX;
    bf16* wkb = wqb + nW;
    bf16* wvb = wkb + nW;
    bf16* wob = wvb + nW;
    bf16* Qb  = wob + nW;           // Q  [B,H,S,64]
    bf16* Kb  = Qb  + nX;           // K  [B,H,S,64]
    bf16* Vtb = Kb  + nX;           // V^T [B,H,64,S]
    bf16* Cb  = Vtb + nX;           // context [B,S,D]
    // total ws use: ~48 MB

    cvt_f32_bf16<<<dim3((unsigned)(nX / 4 / 256)), 256, 0, stream>>>(x,  xb,  (int)(nX / 4));
    cvt_f32_bf16<<<dim3((unsigned)(nW / 4 / 256)), 256, 0, stream>>>(wq, wqb, (int)(nW / 4));
    cvt_f32_bf16<<<dim3((unsigned)(nW / 4 / 256)), 256, 0, stream>>>(wk, wkb, (int)(nW / 4));
    cvt_f32_bf16<<<dim3((unsigned)(nW / 4 / 256)), 256, 0, stream>>>(wv, wvb, (int)(nW / 4));
    cvt_f32_bf16<<<dim3((unsigned)(nW / 4 / 256)), 256, 0, stream>>>(wo, wob, (int)(nW / 4));

    const dim3 gg(D_DIM / 64, (2 * S_LEN) / 128);  // (16, 32)
    gemm_bf16_wmma<<<gg, 256, 0, stream>>>(xb, wqb, Qb,  nullptr, 0);
    gemm_bf16_wmma<<<gg, 256, 0, stream>>>(xb, wkb, Kb,  nullptr, 0);
    gemm_bf16_wmma<<<gg, 256, 0, stream>>>(xb, wvb, Vtb, nullptr, 1);

    attn_fwd_wmma<<<dim3(S_LEN / 64, NH, 2), 128, 0, stream>>>(Qb, Kb, Vtb, Cb);

    gemm_bf16_wmma<<<gg, 256, 0, stream>>>(Cb, wob, nullptr, (float*)d_out, 2);
}
